// Attention_74586402062589
// MI455X (gfx1250) — matrix operand order repaired and checked by reading, not compiled, not run on hardware
//
#include <hip/hip_runtime.h>
#include <hip/hip_bf16.h>

typedef __attribute__((ext_vector_type(16))) __bf16 v16bf;
typedef __attribute__((ext_vector_type(8)))  float  v8f;
typedef __attribute__((ext_vector_type(4)))  unsigned int v4u;
typedef __attribute__((ext_vector_type(8)))  int v8i;
typedef __attribute__((ext_vector_type(4)))  int v4i;

#define WMMA_BF16(a, b, c) \
  __builtin_amdgcn_wmma_f32_16x16x32_bf16(false, (a), false, (b), (short)0, (c), false, false)

union FragU { v16bf v; uint4 q[2]; unsigned short s[16]; };

__device__ inline unsigned short f2bf(float f) {
  unsigned int u = __float_as_uint(f);
  u += 0x7fffu + ((u >> 16) & 1u);          // round-to-nearest-even
  return (unsigned short)(u >> 16);
}
__device__ inline float sigmoidf(float x) { return 1.0f / (1.0f + __expf(-x)); }

// A operand: source row-major [M][K], lane&15 = M. K runs: {c*32+8*hi .. +8} and {+16 .. +8}
__device__ inline v16bf load_a(const unsigned short* base, int stride, int lane, int kc) {
  int col = kc * 32 + ((lane & 16) ? 8 : 0);
  const unsigned short* p = base + (lane & 15) * stride + col;
  FragU f; f.q[0] = *(const uint4*)(p); f.q[1] = *(const uint4*)(p + 16);
  return f.v;
}
// B operand: source is B^T row-major [N][K], lane&15 = N. K runs: {c*32+8*hi .. +8} and {+16 .. +8}
__device__ inline v16bf load_b(const unsigned short* base, int stride, int lane, int kc) {
  int col = kc * 32 + ((lane & 16) ? 8 : 0);
  const unsigned short* p = base + (lane & 15) * stride + col;
  FragU f; f.q[0] = *(const uint4*)(p); f.q[1] = *(const uint4*)(p + 16);
  return f.v;
}

// ---- Tensor Data Mover: 2D bf16 tile (tile_d0 x tile_d1) -> LDS, with row padding ----
// D# per CDNA5 ISA ch.8: group0 = {flags, lds_addr, global_addr, type=2},
// group1 = {data_size=2B + pad cfg, tensor dims, tile dims, dim0 stride}.
// pad_interval code 4 (=32 DWORDs = one 64-elem bf16 row), pad_amount code 3 (=4 DWORDs)
// => LDS row stride 72 bf16 elements (bank-conflict-avoiding), done by the DMA engine.
__device__ inline void tdm_load_2d(unsigned lds_off, const void* gptr,
                                   unsigned tile_d0, unsigned tile_d1,
                                   unsigned tensor_d0, unsigned tensor_d1,
                                   unsigned stride0) {
  unsigned long long ga = (unsigned long long)(uintptr_t)gptr;
  v4u g0;
  g0.x = 1u;                                                  // count=1, no gather
  g0.y = lds_off;                                             // LDS byte address
  g0.z = (unsigned)ga;                                        // global_addr[31:0]
  g0.w = ((unsigned)(ga >> 32) & 0x01FFFFFFu) | (2u << 30);   // addr[56:32] | type=2
  v8i g1;
  g1[0] = (int)((1u << 16) | (1u << 20) | (4u << 22) | (3u << 25)); // 2B, pad on, 32dw/4dw
  g1[1] = (int)((tensor_d0 & 0xFFFFu) << 16);                 // tensor_dim0[15:0]
  g1[2] = (int)((tensor_d0 >> 16) | ((tensor_d1 & 0xFFFFu) << 16));
  g1[3] = (int)((tensor_d1 >> 16) | (tile_d0 << 16));         // tile_dim0
  g1[4] = (int)(tile_d1 & 0xFFFFu);                           // tile_dim1 (tile_dim2=0)
  g1[5] = (int)stride0;                                       // tensor_dim0_stride[31:0]
  g1[6] = 0;
  g1[7] = 0;
  v4i z4 = {0, 0, 0, 0};
#if __clang_major__ >= 23
  v8i z8 = {0, 0, 0, 0, 0, 0, 0, 0};
  __builtin_amdgcn_tensor_load_to_lds(g0, g1, z4, z4, z8, 0);
#else
  __builtin_amdgcn_tensor_load_to_lds(g0, g1, z4, z4, 0);
#endif
}

// ---------------- Kernel 0: transpose + bf16-convert MLP weights into ws ----------------
__global__ __launch_bounds__(256) void prep_w(
    const float* __restrict__ qW1, const float* __restrict__ kW1, const float* __restrict__ vW1,
    const float* __restrict__ qW2, const float* __restrict__ kW2, const float* __restrict__ vW2,
    unsigned short* __restrict__ W1T, unsigned short* __restrict__ W2T) {
  int t = blockIdx.x * 256 + threadIdx.x;
  if (t >= 3 * 4096) return;
  int m = t >> 12, rem = t & 4095, n = rem & 15, k = rem >> 4;
  const float* W1 = (m == 0) ? qW1 : ((m == 1) ? kW1 : vW1);
  const float* W2 = (m == 0) ? qW2 : ((m == 1) ? kW2 : vW2);
  W1T[m * 4096 + n * 256 + k] = f2bf(W1[k * 16 + n]);  // [m][n][k]
  W2T[m * 4096 + n * 256 + k] = f2bf(W2[k * 16 + n]);
}

// ---------------- Kernel A: fused conv3x3 + 3x(two-stage MLP) per (b, ct) ----------------
// grid = (1024, 2), block = 256 (8 waves). The 537MB conv activation never hits HBM.
__global__ __launch_bounds__(256) void conv_mlp(
    const float* __restrict__ x, const float* __restrict__ conv_w, const float* __restrict__ conv_b,
    const float* __restrict__ qb1, const float* __restrict__ kb1, const float* __restrict__ vb1,
    const float* __restrict__ qb2, const float* __restrict__ kb2, const float* __restrict__ vb2,
    const unsigned short* __restrict__ W1T, const unsigned short* __restrict__ W2T,
    unsigned short* __restrict__ qws, unsigned short* __restrict__ kws, unsigned short* __restrict__ vws) {
  __shared__ __align__(16) unsigned short convT[64 * 264];    // 64 conv rows x 256 cols bf16 (padded)
  __shared__ __align__(16) unsigned short s1T[3][16 * 264];   // stage1^T per MLP: [w'][h] bf16

  const int ct = blockIdx.x, b = blockIdx.y;
  const int tid = threadIdx.x, lane = tid & 31, wv = tid >> 5;
  const float* xb = x + b * 4 * 65536;

  float wgt[36];
  #pragma unroll
  for (int i = 0; i < 36; ++i) wgt[i] = conv_w[ct * 36 + i];
  const float bias = conv_b[ct];

  const int r  = tid >> 2;         // conv row 0..63 in tile
  const int wb = (tid & 3) << 6;   // col base 0/64/128/192

  for (int h0 = 0; h0 < 256; h0 += 64) {
    // ---- conv: each thread 64 outputs of one row, sliding 3-tap window ----
    float acc[64];
    #pragma unroll
    for (int j = 0; j < 64; ++j) acc[j] = 0.f;
    const int h = h0 + r;
    #pragma unroll
    for (int c = 0; c < 4; ++c) {
      const float* xc = xb + c * 65536;
      #pragma unroll
      for (int dh = 0; dh < 3; ++dh) {
        int hh = h + dh - 1;
        if (hh >= 0 && hh <= 255) {
          const float* row = xc + hh * 256 + wb;
          if (h0 + 64 < 256) __builtin_prefetch(row + 64 * 256, 0, 1);  // next h-tile
          float w0 = wgt[(c * 3 + dh) * 3 + 0];
          float w1 = wgt[(c * 3 + dh) * 3 + 1];
          float w2 = wgt[(c * 3 + dh) * 3 + 2];
          float prev = (wb > 0) ? row[-1] : 0.f;
          float cur  = row[0];
          #pragma unroll
          for (int j = 0; j < 64; ++j) {
            float nxt = (wb + j + 1 < 256) ? row[j + 1] : 0.f;
            acc[j] = fmaf(prev, w0, fmaf(cur, w1, fmaf(nxt, w2, acc[j])));
            prev = cur; cur = nxt;
          }
        }
      }
    }
    unsigned short* dst = &convT[r * 264 + wb];
    #pragma unroll
    for (int j = 0; j < 64; ++j) dst[j] = f2bf(acc[j] + bias);
    __syncthreads();

    // ---- stage1 WMMA: 12 chains = 3 MLPs x 4 M-tiles; D stored transposed into s1T ----
    for (int chain = wv; chain < 12; chain += 8) {
      int m = chain >> 2, mt = chain & 3;
      v8f d = {0.f, 0.f, 0.f, 0.f, 0.f, 0.f, 0.f, 0.f};
      #pragma unroll
      for (int kc = 0; kc < 8; ++kc) {
        v16bf A  = load_a(&convT[mt * 16 * 264], 264, lane, kc);
        v16bf Bf = load_b(W1T + m * 4096, 256, lane, kc);
        d = WMMA_BF16(A, Bf, d);
      }
      const float* b1p = (m == 0) ? qb1 : ((m == 1) ? kb1 : vb1);
      float b1v = b1p[lane & 15];
      FragU out;
      #pragma unroll
      for (int rr = 0; rr < 8; ++rr) out.s[rr] = f2bf(sigmoidf(d[rr] + b1v));
      int hloc = h0 + mt * 16 + ((lane >> 4) << 3);            // contiguous h per lane
      *(uint4*)(&s1T[m][(lane & 15) * 264 + hloc]) = out.q[0]; // transposed b128 store
    }
    __syncthreads();
  }

  // ---- stage2 WMMA (16x16, K=256): A = W2^T, B = s1 (via s1T); then head-split store ----
  if (wv < 3) {
    int m = wv;
    v8f d = {0.f, 0.f, 0.f, 0.f, 0.f, 0.f, 0.f, 0.f};
    #pragma unroll
    for (int kc = 0; kc < 8; ++kc) {
      v16bf A  = load_a(W2T + m * 4096, 256, lane, kc);
      v16bf Bf = load_b(&s1T[m][0], 264, lane, kc);
      d = WMMA_BF16(A, Bf, d);
    }
    const float* b2p = (m == 0) ? qb2 : ((m == 1) ? kb2 : vb2);
    #pragma unroll
    for (int rr = 0; rr < 8; ++rr) {
      int hp = rr + ((lane >> 4) << 3);   // h' (row)
      int wp = lane & 15;                 // w' (col)
      float val = sigmoidf(d[rr] + b2p[hp]);
      int n   = ((hp & 1) << 1) | (wp & 1);          // head = (h'%2)*2 + w'%2
      int idx = ((hp >> 1) << 3) | (wp >> 1);        // (h'/2)*8 + w'/2
      unsigned short bv = f2bf(val);
      if (m == 0)      qws[(((b * 4 + n) * 1024 + ct) << 6) + idx] = bv;  // [bn][ct][64]
      else if (m == 1) kws[(((b * 4 + n) * 1024 + ct) << 6) + idx] = bv;  // [bn][ct][64]
      else             vws[(((b * 4 + n) << 6) + idx) * 1024 + ct] = bv;  // v^T: [bn][64][ct]
    }
  }
}

// ---------------- Kernel B: flash attention per (b,n), 128 q-rows per WG ----------------
// grid = (8 mblocks, 8 bn), block = 256 (8 waves; wave owns 16 q-rows).
// K/V 64x64 bf16 tiles are DMA'd into LDS once per WG by the Tensor Data Mover
// (instead of 8 waves redundantly loading identical B fragments from global).
__global__ __launch_bounds__(256) void attention(
    const unsigned short* __restrict__ qws, const unsigned short* __restrict__ kws,
    const unsigned short* __restrict__ vws, const float* __restrict__ temperature,
    float* __restrict__ out) {
  __shared__ __align__(16) unsigned short Ps[8][16 * 72];  // per-wave softmax tile (bf16)
  __shared__ __align__(16) unsigned short Ks[64 * 72];     // TDM-padded K tile [e][x]
  __shared__ __align__(16) unsigned short Vs[64 * 72];     // TDM-padded v^T tile [x][e]

  const int tid = threadIdx.x, lane = tid & 31, wv = tid >> 5;
  const int bn = blockIdx.y;            // b*4 + n
  const int m0 = blockIdx.x * 128;
  const float tv = temperature[bn & 3];

  const unsigned short* qrow = qws + (bn * 1024 + m0 + wv * 16) * 64;
  v16bf qa0 = load_a(qrow, 64, lane, 0);
  v16bf qa1 = load_a(qrow, 64, lane, 1);

  float mrow[8], lsum[8];
  v8f O[4];
  const v8f vz = {0.f, 0.f, 0.f, 0.f, 0.f, 0.f, 0.f, 0.f};
  #pragma unroll
  for (int nt = 0; nt < 4; ++nt) O[nt] = vz;
  #pragma unroll
  for (int rr = 0; rr < 8; ++rr) { mrow[rr] = -3.0e38f; lsum[rr] = 0.f; }

  unsigned short* myP = &Ps[wv][0];
  const int prow = (lane >> 4) << 3;
  const unsigned ksOff = (unsigned)(uintptr_t)&Ks[0];
  const unsigned vsOff = (unsigned)(uintptr_t)&Vs[0];

  for (int e0 = 0; e0 < 1024; e0 += 64) {
    // ---- TDM: one wave DMAs the K tile (rows e0..e0+63 of kws[bn]) and the
    //      v^T tile (cols e0..e0+63 of vws[bn]) into padded LDS ----
    if (wv == 0) {
      tdm_load_2d(ksOff, kws + (bn * 1024 + e0) * 64, 64, 64, 64, 1024, 64);
      tdm_load_2d(vsOff, vws + bn * 64 * 1024 + e0, 64, 64, 1024, 64, 1024);
      __builtin_amdgcn_s_wait_tensorcnt(0);
    }
    __syncthreads();

    // scores tiles S = (q @ k^T) * tau, K fragments from LDS
    v8f S[4];
    #pragma unroll
    for (int nt = 0; nt < 4; ++nt) {
      v8f a = vz;
      const unsigned short* krow = &Ks[nt * 16 * 72];
      a = WMMA_BF16(qa0, load_b(krow, 72, lane, 0), a);
      a = WMMA_BF16(qa1, load_b(krow, 72, lane, 1), a);
      #pragma unroll
      for (int rr = 0; rr < 8; ++rr) a[rr] *= tv;
      S[nt] = a;
    }
    // online softmax: row max over 4 tiles + 16 lanes of the half
    float nm[8];
    #pragma unroll
    for (int rr = 0; rr < 8; ++rr) {
      float v = mrow[rr];
      #pragma unroll
      for (int nt = 0; nt < 4; ++nt) v = fmaxf(v, S[nt][rr]);
      nm[rr] = v;
    }
    #pragma unroll
    for (int off = 8; off >= 1; off >>= 1) {
      #pragma unroll
      for (int rr = 0; rr < 8; ++rr) nm[rr] = fmaxf(nm[rr], __shfl_xor(nm[rr], off, 32));
    }
    float ps[8];
    #pragma unroll
    for (int rr = 0; rr < 8; ++rr) {
      float sc = __expf(mrow[rr] - nm[rr]);
      lsum[rr] *= sc;
      #pragma unroll
      for (int nt = 0; nt < 4; ++nt) O[nt][rr] *= sc;
      mrow[rr] = nm[rr];
      ps[rr] = 0.f;
    }
    // p = exp(S - m); transpose through LDS so p can be a WMMA A operand
    #pragma unroll
    for (int nt = 0; nt < 4; ++nt) {
      #pragma unroll
      for (int rr = 0; rr < 8; ++rr) {
        float p = __expf(S[nt][rr] - nm[rr]);
        ps[rr] += p;
        myP[(prow + rr) * 72 + nt * 16 + (lane & 15)] = f2bf(p);
      }
    }
    #pragma unroll
    for (int off = 8; off >= 1; off >>= 1) {
      #pragma unroll
      for (int rr = 0; rr < 8; ++rr) ps[rr] += __shfl_xor(ps[rr], off, 32);
    }
    #pragma unroll
    for (int rr = 0; rr < 8; ++rr) lsum[rr] += ps[rr];

    v16bf pa0 = load_a(myP, 72, lane, 0);
    v16bf pa1 = load_a(myP, 72, lane, 1);
    #pragma unroll
    for (int nt = 0; nt < 4; ++nt) {
      const unsigned short* vrow = &Vs[nt * 16 * 72];
      O[nt] = WMMA_BF16(pa0, load_b(vrow, 72, lane, 0), O[nt]);
      O[nt] = WMMA_BF16(pa1, load_b(vrow, 72, lane, 1), O[nt]);
    }
    __syncthreads();   // before next iteration's DMA overwrites Ks/Vs
  }
  // normalize + store (output layout == final reshape: flat (bn, ct, 64))
  #pragma unroll
  for (int rr = 0; rr < 8; ++rr) {
    float inv = 1.0f / lsum[rr];
    int row = m0 + wv * 16 + prow + rr;
    #pragma unroll
    for (int nt = 0; nt < 4; ++nt)
      out[(bn * 1024 + row) * 64 + nt * 16 + (lane & 15)] = O[nt][rr] * inv;
  }
}

extern "C" void kernel_launch(void* const* d_in, const int* in_sizes, int n_in,
                              void* d_out, int out_size, void* d_ws, size_t ws_size,
                              hipStream_t stream) {
  const float* x      = (const float*)d_in[0];
  const float* conv_w = (const float*)d_in[1];
  const float* conv_b = (const float*)d_in[2];
  const float* qW1 = (const float*)d_in[3],  *qb1 = (const float*)d_in[4];
  const float* qW2 = (const float*)d_in[5],  *qb2 = (const float*)d_in[6];
  const float* kW1 = (const float*)d_in[7],  *kb1 = (const float*)d_in[8];
  const float* kW2 = (const float*)d_in[9],  *kb2 = (const float*)d_in[10];
  const float* vW1 = (const float*)d_in[11], *vb1 = (const float*)d_in[12];
  const float* vW2 = (const float*)d_in[13], *vb2 = (const float*)d_in[14];
  const float* temperature = (const float*)d_in[15];
  float* out = (float*)d_out;

  char* ws = (char*)d_ws;
  unsigned short* W1T = (unsigned short*)(ws);                        // 3*16*256 bf16
  unsigned short* W2T = (unsigned short*)(ws + 24576);
  unsigned short* qws = (unsigned short*)(ws + 49152);                // [2*4][1024][64] bf16
  unsigned short* kws = (unsigned short*)(ws + 49152 + 1048576);
  unsigned short* vws = (unsigned short*)(ws + 49152 + 2 * 1048576);  // v^T: [2*4][64][1024]

  prep_w<<<48, 256, 0, stream>>>(qW1, kW1, vW1, qW2, kW2, vW2, W1T, W2T);
  dim3 gA(1024, 2);
  conv_mlp<<<gA, 256, 0, stream>>>(x, conv_w, conv_b, qb1, kb1, vb1, qb2, kb2, vb2,
                                   W1T, W2T, qws, kws, vws);
  dim3 gB(8, 8);
  attention<<<gB, 256, 0, stream>>>(qws, kws, vws, temperature, out);
}
